// SimpleAttention_40656160424543
// MI455X (gfx1250) — matrix operand order repaired and checked by reading, not compiled, not run on hardware
//
#include <hip/hip_runtime.h>

// Problem constants (match reference)
#define B_  2
#define S_  2048
#define D_  1024
#define H_  16
#define HD_ 64
#define SCALE_ 0.125f   // 1/sqrt(64)

typedef __bf16 bf16;
typedef __attribute__((ext_vector_type(8)))  __bf16 v8bf;
typedef __attribute__((ext_vector_type(16))) __bf16 v16bf;
typedef __attribute__((ext_vector_type(8)))  float  v8f;
typedef __attribute__((ext_vector_type(4)))  unsigned int v4u;
typedef __attribute__((ext_vector_type(4)))  int v4i;
typedef __attribute__((ext_vector_type(8)))  int v8i;

#define WMMA_BF16(A_, B_m, C_) \
  __builtin_amdgcn_wmma_f32_16x16x32_bf16(false, (A_), false, (B_m), (short)0, (C_), false, false)

#if defined(__HIP_DEVICE_COMPILE__) && __has_builtin(__builtin_amdgcn_tensor_load_to_lds) && \
    __has_builtin(__builtin_amdgcn_s_wait_tensorcnt)
#define HAVE_TDM 1
#else
#define HAVE_TDM 0
#endif

// ---------------------------------------------------------------------------
// Fragment loaders (wave32).
// A-matrix 16x32 bf16 (MxK): lane L (L<16) holds row M=L, K {0..7,16..23};
// lane L+16 holds row M=L, K {8..15,24..31}. Two 16-byte contiguous loads.
static __device__ __forceinline__ v16bf frag_a(const bf16* base, int ld, int lane) {
  const int r  = lane & 15;
  const int ko = (lane >> 4) << 3;          // 0 or 8
  const bf16* p = base + (size_t)r * ld + ko;
  v8bf lo = *(const v8bf*)p;
  v8bf hi = *(const v8bf*)(p + 16);
  return __builtin_shufflevector(lo, hi, 0,1,2,3,4,5,6,7,8,9,10,11,12,13,14,15);
}

// B-matrix 32x16 bf16 (KxN), loaded from row-major B^T [N][K]:
// lane L holds column N=L&15, K {(L>>4)*8 .. +7, 16+(L>>4)*8 .. +7}.
static __device__ __forceinline__ v16bf frag_b(const bf16* base, int ld, int lane) {
  const int n  = lane & 15;
  const int ko = (lane >> 4) << 3;          // 0 or 8
  const bf16* p = base + (size_t)n * ld + ko;
  v8bf lo = *(const v8bf*)p;
  v8bf hi = *(const v8bf*)(p + 16);
  return __builtin_shufflevector(lo, hi, 0,1,2,3,4,5,6,7,8,9,10,11,12,13,14,15);
}

// ---------------------------------------------------------------------------
// Tensor Data Mover: 2D tile load Global -> LDS (bf16 elements).
// Descriptor layout per CDNA5 ISA ch.8 (D# group0/group1); groups 2/3 zero
// (2D tensor, tile_dim2=0 => unused).
#if HAVE_TDM
static __device__ __forceinline__ void tdm_load_2d(
    void* lds_dst, const void* gsrc,
    unsigned tensor_d0, unsigned tensor_d1,          // tensor extents (elems)
    unsigned long long stride0,                      // elems between dim1 lines
    unsigned tile_d0, unsigned tile_d1)              // tile extents (elems)
{
  const unsigned long long ga = (unsigned long long)(size_t)gsrc;
  const unsigned lds = (unsigned)(size_t)lds_dst;    // low 32 bits = LDS offset

  v4u g0;
  g0[0] = 1u;                                            // count=1 (valid user D#)
  g0[1] = lds;                                           // lds_addr
  g0[2] = (unsigned)(ga & 0xFFFFFFFFu);                  // global_addr[31:0]
  g0[3] = (unsigned)((ga >> 32) & 0x1FFFFFFu) | (2u << 30); // addr[56:32] | type=2

  v8i g1;
  g1[0] = (int)(1u << 16);                               // data_size=1 -> 2 bytes
  g1[1] = (int)((tensor_d0 & 0xFFFFu) << 16);            // tensor_dim0[15:0]
  g1[2] = (int)((tensor_d0 >> 16) | ((tensor_d1 & 0xFFFFu) << 16));
  g1[3] = (int)((tensor_d1 >> 16) | (tile_d0 << 16));    // tensor_dim1 hi | tile_dim0
  g1[4] = (int)(tile_d1 & 0xFFFFu);                      // tile_dim1 | tile_dim2=0
  g1[5] = (int)(unsigned)(stride0 & 0xFFFFFFFFu);        // tensor_dim0_stride lo
  g1[6] = (int)(unsigned)((stride0 >> 32) & 0xFFFFu);    // stride0 hi | dim1_stride=0
  g1[7] = 0;

  const v4i z4 = (v4i){0, 0, 0, 0};
#if __clang_major__ >= 23
  const v8i z8 = (v8i){0, 0, 0, 0, 0, 0, 0, 0};
  __builtin_amdgcn_tensor_load_to_lds(g0, g1, z4, z4, z8, 0);
#else
  __builtin_amdgcn_tensor_load_to_lds(g0, g1, z4, z4, 0);
#endif
}
#endif

// ---------------------------------------------------------------------------
// Stage 0: f32 -> bf16 conversion (grid sized exactly, no bounds check)
__global__ void sattn_cvt_bf16(const float* __restrict__ in, bf16* __restrict__ out) {
  const int i = blockIdx.x * blockDim.x + threadIdx.x;
  out[i] = (bf16)in[i];
}

// ---------------------------------------------------------------------------
// Shared GEMM body: C[16 x 64 per wave] = A[16 x 1024] * B^T[N x 1024]^T.
// All 8 waves of a block share one A tile (TDM-staged in LDS); the K-loop is
// register double-buffered so fragment loads overlap the 4 WMMAs.
// ---------------------------------------------------------------------------
// Stage 1: QKV projection.  out[m,e] = sum_d x[m,d] * w_qkv[e,d]
// Epilogue scatters into Q[b,h,s,hd], K[b,h,s,hd], Vt[b,h,hd,s] (bf16).
__global__ void __launch_bounds__(256)
sattn_gemm_qkv(const bf16* __restrict__ Xb, const bf16* __restrict__ Wb,
               bf16* __restrict__ Qb, bf16* __restrict__ Kb, bf16* __restrict__ Vt) {
  __shared__ __align__(16) bf16 ldsA[16][D_];            // 32 KB A tile
  const int lane = threadIdx.x & 31;
  const int w    = threadIdx.x >> 5;
  const int m0   = (blockIdx.x / 6) * 16;                // 6 n-groups of 8 waves
  const int n0   = ((blockIdx.x % 6) * 8 + w) * 64;
  const bf16* arow = Xb + (size_t)m0 * D_;

#if HAVE_TDM
  if (w == 0) {
    tdm_load_2d(&ldsA[0][0], arow, D_, 16, D_, D_, 16);  // contiguous 16x1024 tile
    __builtin_amdgcn_s_wait_tensorcnt(0);
  }
  __syncthreads();
  const bf16* aptr = &ldsA[0][0];
#else
  const bf16* aptr = arow;
#endif

  v8f acc[4];
#pragma unroll
  for (int j = 0; j < 4; ++j) acc[j] = (v8f){0.f,0.f,0.f,0.f,0.f,0.f,0.f,0.f};

  v16bf va = frag_a(aptr, D_, lane);
  v16bf vb[4];
#pragma unroll
  for (int j = 0; j < 4; ++j) vb[j] = frag_b(Wb + (size_t)(n0 + 16 * j) * D_, D_, lane);

#pragma unroll 4
  for (int k0 = 0; k0 < D_ - 32; k0 += 32) {
    v16bf van = frag_a(aptr + k0 + 32, D_, lane);        // prefetch next k-step
    v16bf vbn[4];
#pragma unroll
    for (int j = 0; j < 4; ++j)
      vbn[j] = frag_b(Wb + (size_t)(n0 + 16 * j) * D_ + k0 + 32, D_, lane);
#pragma unroll
    for (int j = 0; j < 4; ++j) acc[j] = WMMA_BF16(va, vb[j], acc[j]);
    va = van;
#pragma unroll
    for (int j = 0; j < 4; ++j) vb[j] = vbn[j];
  }
#pragma unroll
  for (int j = 0; j < 4; ++j) acc[j] = WMMA_BF16(va, vb[j], acc[j]);

  // C/D layout: VGPR r, lanes 0-15 -> row r, lanes 16-31 -> row r+8; col=lane&15
  const int col  = lane & 15;
  const int roff = (lane >> 4) << 3;
#pragma unroll
  for (int j = 0; j < 4; ++j) {
#pragma unroll
    for (int r = 0; r < 8; ++r) {
      const int m = m0 + roff + r;
      const int e = n0 + j * 16 + col;
      const int b = m >> 11, s = m & (S_ - 1);
      const int which = e >> 10, rem = e & 1023;
      const int h = rem >> 6, hd = rem & 63;
      const size_t bh = (size_t)b * H_ + h;
      const bf16 v = (bf16)acc[j][r];
      if      (which == 0) Qb[(bh * S_ + s) * HD_ + hd] = v;
      else if (which == 1) Kb[(bh * S_ + s) * HD_ + hd] = v;
      else                 Vt[(bh * HD_ + hd) * S_ + s] = v;   // V transposed
    }
  }
}

// ---------------------------------------------------------------------------
// Stage 2: flash attention. One wave per 16 query rows; 4 waves / block share
// (b,h). K/V tiles are TDM-loaded into double-buffered LDS: wave 0 issues the
// next tile pair BEFORE waiting (s_wait_tensorcnt 2 + in-order completion =>
// current pair has landed), overlapping DMA with compute.
__global__ void __launch_bounds__(128)
sattn_flash(const bf16* __restrict__ Qb, const bf16* __restrict__ Kb,
            const bf16* __restrict__ Vt, bf16* __restrict__ Ob) {
  __shared__ __align__(16) bf16 ldsK[2][32][HD_];        // 8 KB, [s-local][hd]
  __shared__ __align__(16) bf16 ldsV[2][HD_][32];        // 8 KB, [hd][s-local]
  __shared__ __align__(16) bf16 pbuf[4][16][32];         // 4 KB P re-layout

  const int lane = threadIdx.x & 31;
  const int w    = threadIdx.x >> 5;
  const int gw   = blockIdx.x * 4 + w;                   // 0 .. B*H*S/16-1
  const int bh   = gw >> 7;                              // 128 q-tiles per (b,h)
  const int q0   = (gw & 127) << 4;

  const bf16* Qbase = Qb + ((size_t)bh * S_ + q0) * HD_;
  const bf16* Kbase = Kb + (size_t)bh * S_ * HD_;
  const bf16* Vbase = Vt + (size_t)bh * HD_ * S_;

  const v16bf aq0 = frag_a(Qbase,      HD_, lane);       // hd 0..31
  const v16bf aq1 = frag_a(Qbase + 32, HD_, lane);       // hd 32..63

  v8f o[4];
#pragma unroll
  for (int j = 0; j < 4; ++j) o[j] = (v8f){0.f,0.f,0.f,0.f,0.f,0.f,0.f,0.f};
  float mrow[8], lrow[8];
#pragma unroll
  for (int r = 0; r < 8; ++r) { mrow[r] = -3.0e38f; lrow[r] = 0.f; }

  const int col  = lane & 15;
  const int half = lane >> 4;
  const int NIT  = S_ / 32;                              // 64

#if HAVE_TDM
  if (w == 0) {                                          // prologue: tile 0
    tdm_load_2d(&ldsK[0][0][0], Kbase, HD_, 32, HD_, HD_, 32);
    tdm_load_2d(&ldsV[0][0][0], Vbase, 32, HD_, S_, 32, HD_);
  }
#endif

  for (int it = 0; it < NIT; ++it) {
    const int cur = it & 1;
    const int sk  = it * 32;

#if HAVE_TDM
    if (w == 0) {
      if (it < NIT - 1) {                                // issue next, then wait
        tdm_load_2d(&ldsK[cur ^ 1][0][0], Kbase + (size_t)(sk + 32) * HD_,
                    HD_, 32, HD_, HD_, 32);
        tdm_load_2d(&ldsV[cur ^ 1][0][0], Vbase + (sk + 32),
                    32, HD_, S_, 32, HD_);
        __builtin_amdgcn_s_wait_tensorcnt(2);            // oldest pair done
      } else {
        __builtin_amdgcn_s_wait_tensorcnt(0);
      }
    }
    __syncthreads();                                     // tile `cur` visible
    const bf16* kt = &ldsK[cur][0][0];  const int ldk = HD_;
    const bf16* vt = &ldsV[cur][0][0];  const int ldv = 32;
#else
    const bf16* kt = Kbase + (size_t)sk * HD_;  const int ldk = HD_;
    const bf16* vt = Vbase + sk;                const int ldv = S_;
#endif

    // scores S = Q K^T : two 16x16 n-tiles, HD=64 -> two K=32 chunks each
    const v16bf bk00 = frag_b(kt,                 ldk, lane);
    const v16bf bk01 = frag_b(kt + 32,            ldk, lane);
    const v16bf bk10 = frag_b(kt + 16 * ldk,      ldk, lane);
    const v16bf bk11 = frag_b(kt + 16 * ldk + 32, ldk, lane);

    const v8f z = (v8f){0.f,0.f,0.f,0.f,0.f,0.f,0.f,0.f};
    v8f s0 = WMMA_BF16(aq0, bk00, z);
    s0     = WMMA_BF16(aq1, bk01, s0);
    v8f s1 = WMMA_BF16(aq0, bk10, z);
    s1     = WMMA_BF16(aq1, bk11, s1);

    float alpha[8], p0[8], p1[8];
#pragma unroll
    for (int r = 0; r < 8; ++r) {
      const float a = s0[r] * SCALE_;
      const float b = s1[r] * SCALE_;
      float t = fmaxf(a, b);
#pragma unroll
      for (int msk = 1; msk <= 8; msk <<= 1) t = fmaxf(t, __shfl_xor(t, msk, 16));
      const float mnew = fmaxf(mrow[r], t);
      alpha[r] = __expf(mrow[r] - mnew);
      p0[r]    = __expf(a - mnew);
      p1[r]    = __expf(b - mnew);
      float sum = p0[r] + p1[r];
#pragma unroll
      for (int msk = 1; msk <= 8; msk <<= 1) sum += __shfl_xor(sum, msk, 16);
      lrow[r] = lrow[r] * alpha[r] + sum;
      mrow[r] = mnew;
    }

    // C-layout -> A-fragment layout via LDS
#pragma unroll
    for (int r = 0; r < 8; ++r) {
      pbuf[w][(half << 3) + r][col]      = (bf16)p0[r];
      pbuf[w][(half << 3) + r][col + 16] = (bf16)p1[r];
    }
    __syncthreads();
    const v16bf pa = frag_a(&pbuf[w][0][0], 32, lane);
    __syncthreads();

    // O = diag(alpha)*O + P * V
#pragma unroll
    for (int j = 0; j < 4; ++j) {
#pragma unroll
      for (int r = 0; r < 8; ++r) o[j][r] *= alpha[r];
      const v16bf bv = frag_b(vt + (size_t)(j * 16) * ldv, ldv, lane);
      o[j] = WMMA_BF16(pa, bv, o[j]);
    }

#if HAVE_TDM
    __syncthreads();   // all waves done reading buffer `cur` before TDM reuse
#endif
  }

  const int b = bh >> 4, h = bh & 15;
#pragma unroll
  for (int j = 0; j < 4; ++j) {
#pragma unroll
    for (int r = 0; r < 8; ++r) {
      const int row = q0 + (half << 3) + r;
      const int d   = h * HD_ + j * 16 + col;
      Ob[((size_t)b * S_ + row) * D_ + d] = (bf16)(o[j][r] / lrow[r]);
    }
  }
}

// ---------------------------------------------------------------------------
// Stage 3: output projection. out[m,n] = sum_d O[m,d] * w_out[n,d], f32 out.
__global__ void __launch_bounds__(256)
sattn_gemm_out(const bf16* __restrict__ Ab, const bf16* __restrict__ Wb,
               float* __restrict__ C) {
  __shared__ __align__(16) bf16 ldsA[16][D_];            // 32 KB A tile
  const int lane = threadIdx.x & 31;
  const int w    = threadIdx.x >> 5;
  const int m0   = (blockIdx.x / 2) * 16;                // 2 n-groups of 8 waves
  const int n0   = ((blockIdx.x % 2) * 8 + w) * 64;
  const bf16* arow = Ab + (size_t)m0 * D_;

#if HAVE_TDM
  if (w == 0) {
    tdm_load_2d(&ldsA[0][0], arow, D_, 16, D_, D_, 16);
    __builtin_amdgcn_s_wait_tensorcnt(0);
  }
  __syncthreads();
  const bf16* aptr = &ldsA[0][0];
#else
  const bf16* aptr = arow;
#endif

  v8f acc[4];
#pragma unroll
  for (int j = 0; j < 4; ++j) acc[j] = (v8f){0.f,0.f,0.f,0.f,0.f,0.f,0.f,0.f};

  v16bf va = frag_a(aptr, D_, lane);
  v16bf vb[4];
#pragma unroll
  for (int j = 0; j < 4; ++j) vb[j] = frag_b(Wb + (size_t)(n0 + 16 * j) * D_, D_, lane);

#pragma unroll 4
  for (int k0 = 0; k0 < D_ - 32; k0 += 32) {
    v16bf van = frag_a(aptr + k0 + 32, D_, lane);
    v16bf vbn[4];
#pragma unroll
    for (int j = 0; j < 4; ++j)
      vbn[j] = frag_b(Wb + (size_t)(n0 + 16 * j) * D_ + k0 + 32, D_, lane);
#pragma unroll
    for (int j = 0; j < 4; ++j) acc[j] = WMMA_BF16(va, vb[j], acc[j]);
    va = van;
#pragma unroll
    for (int j = 0; j < 4; ++j) vb[j] = vbn[j];
  }
#pragma unroll
  for (int j = 0; j < 4; ++j) acc[j] = WMMA_BF16(va, vb[j], acc[j]);

  const int col  = lane & 15;
  const int roff = (lane >> 4) << 3;
#pragma unroll
  for (int j = 0; j < 4; ++j)
#pragma unroll
    for (int r = 0; r < 8; ++r)
      C[(size_t)(m0 + roff + r) * D_ + n0 + j * 16 + col] = acc[j][r];
}

// ---------------------------------------------------------------------------
extern "C" void kernel_launch(void* const* d_in, const int* in_sizes, int n_in,
                              void* d_out, int out_size, void* d_ws, size_t ws_size,
                              hipStream_t stream) {
  const float* x    = (const float*)d_in[0];   // [B,S,D]
  const float* wqkv = (const float*)d_in[1];   // [3D,D]
  const float* wout = (const float*)d_in[2];   // [D,D]
  float* out = (float*)d_out;

  // workspace carve (bf16 buffers), total 48 MB
  char* ws = (char*)d_ws;
  const size_t NX  = (size_t)B_ * S_ * D_;        // 4 Mi elems
  const size_t NWQ = (size_t)3 * D_ * D_;         // 3 Mi
  const size_t NWO = (size_t)D_ * D_;             // 1 Mi
  const size_t NQ  = (size_t)B_ * H_ * S_ * HD_;  // 4 Mi

  bf16* x_bf    = (bf16*)ws;                          ws += NX  * 2;
  bf16* wqkv_bf = (bf16*)ws;                          ws += NWQ * 2;
  bf16* wout_bf = (bf16*)ws;                          ws += NWO * 2;
  bf16* Qb      = (bf16*)ws;                          ws += NQ  * 2;
  bf16* Kb      = (bf16*)ws;                          ws += NQ  * 2;
  bf16* Vt      = (bf16*)ws;                          ws += NQ  * 2;
  bf16* Ob      = (bf16*)ws;                          ws += NX  * 2;

  // Stage 0: conversions (exact grids)
  sattn_cvt_bf16<<<(int)(NX  / 256), 256, 0, stream>>>(x,    x_bf);
  sattn_cvt_bf16<<<(int)(NWQ / 256), 256, 0, stream>>>(wqkv, wqkv_bf);
  sattn_cvt_bf16<<<(int)(NWO / 256), 256, 0, stream>>>(wout, wout_bf);

  // Stage 1: QKV gemm. 256 m-tiles * 6 n-groups = 1536 blocks (8 waves each)
  sattn_gemm_qkv<<<1536, 256, 0, stream>>>(x_bf, wqkv_bf, Qb, Kb, Vt);

  // Stage 2: flash attention. B*H*(S/16)=4096 waves, 4 waves/block -> 1024 blocks
  sattn_flash<<<1024, 128, 0, stream>>>(Qb, Kb, Vt, Ob);

  // Stage 3: output gemm. 256 m-tiles * 2 n-groups = 512 blocks
  sattn_gemm_out<<<512, 256, 0, stream>>>(Ob, wout_bf, out);
}
